// Experts_79190607004281
// MI455X (gfx1250) — matrix operand order repaired and checked by reading, not compiled, not run on hardware
//
#include <hip/hip_runtime.h>
#include <math.h>

typedef _Float16 f16;
typedef __attribute__((ext_vector_type(16))) _Float16     v16h;
typedef __attribute__((ext_vector_type(8)))  float        v8f;
typedef __attribute__((ext_vector_type(4)))  unsigned int v4u;
typedef __attribute__((ext_vector_type(8)))  unsigned int v8u;

// Problem constants (B=4, E=8, S=1024, D=1024, H=2048)
static constexpr int NEXP = 8;
static constexpr int TROW = 4096;   // B*S rows per expert
static constexpr int DDIM = 1024;
static constexpr int HDIM = 2048;

// GEMM tiling
static constexpr int BM  = 128;
static constexpr int BN  = 128;
static constexpr int BK  = 64;      // two 16x16x32 K-steps per stage
static constexpr int LDK = 72;      // halves per padded LDS row: 128B data + 16B pad = 144B

// ---------------------------------------------------------------------------
// Prep kernel 1: fp32 -> f16 elementwise (n multiple of 2048)
// ---------------------------------------------------------------------------
__global__ __launch_bounds__(256)
void cvt_f32_to_f16(const float* __restrict__ in, f16* __restrict__ out, long long n)
{
    long long i = ((long long)blockIdx.x * 256 + threadIdx.x) * 8;
    if (i >= n) return;
    const float4* s = (const float4*)(in + i);
    float4 f0 = s[0], f1 = s[1];
    union { f16 h[8]; uint4 q; } t;
    t.h[0] = (f16)f0.x; t.h[1] = (f16)f0.y; t.h[2] = (f16)f0.z; t.h[3] = (f16)f0.w;
    t.h[4] = (f16)f1.x; t.h[5] = (f16)f1.y; t.h[6] = (f16)f1.z; t.h[7] = (f16)f1.w;
    *(uint4*)(out + i) = t.q;
}

// ---------------------------------------------------------------------------
// Prep kernel 2: per-expert fp32 (R x C) -> f16 transposed (C x R), 32x32 tiles
// ---------------------------------------------------------------------------
__global__ __launch_bounds__(256)
void transpose_cvt_f16(const float* __restrict__ in, f16* __restrict__ out, int R, int C)
{
    __shared__ f16 t[32][33];
    const size_t base = (size_t)blockIdx.z * R * C;
    const int c0 = blockIdx.x * 32, r0 = blockIdx.y * 32;
    const int tx = threadIdx.x & 31, ty = threadIdx.x >> 5;
#pragma unroll
    for (int i = 0; i < 4; ++i) {
        const int r = ty + i * 8;
        t[r][tx] = (f16)in[base + (size_t)(r0 + r) * C + (c0 + tx)];
    }
    __syncthreads();
#pragma unroll
    for (int i = 0; i < 4; ++i) {
        const int oc = ty + i * 8;   // column of input == row of output
        out[base + (size_t)(c0 + oc) * R + (r0 + tx)] = t[tx][oc];
    }
}

// ---------------------------------------------------------------------------
// TDM: issue a 2D f16 tile DMA (BM rows x BK halves) global -> LDS.
// D# per CDNA5 ISA ch.8: group0 {count=1, lds_addr, global_addr, type=2},
// group1 {data_size=2B, pad: +4 dwords per 32 dwords -> 144B LDS row pitch,
//         tensor_dim0=K, tile_dim0=BK, tile_dim1=BM, tensor_dim0_stride=K}.
// 2-group (2D) form; tracked by TENSORcnt.
// ---------------------------------------------------------------------------
__device__ __forceinline__ void tdm_tile_load(unsigned lds_off, const f16* gptr, int K)
{
    const unsigned long long ga = (unsigned long long)(size_t)gptr;
    v4u g0;
    g0[0] = 1u;                                   // count=1, user mode, no gather
    g0[1] = lds_off;                              // LDS byte address
    g0[2] = (unsigned)ga;                         // global_addr[31:0]
    g0[3] = (unsigned)(ga >> 32) | (2u << 30);    // global_addr[56:32] | type=2

    const unsigned td0 = (unsigned)K;             // tensor dim0 (row length, elems)
    const unsigned td1 = 1u << 20;                // tensor dim1 (rows) - generous, tile in-bounds
    v8u g1;
    g1[0] = (1u << 16)                            // data_size = 2 bytes
          | (1u << 20)                            // pad_enable
          | (4u << 22)                            // pad_interval: every 32 dwords (=128B row)
          | (3u << 25);                           // pad_amount: 4 dwords (=16B)
    g1[1] = (td0 & 0xFFFFu) << 16;                // tensor_dim0[15:0]
    g1[2] = (td0 >> 16) | ((td1 & 0xFFFFu) << 16);
    g1[3] = (td1 >> 16) | ((unsigned)BK << 16);   // tile_dim0 = 64 elems (128B)
    g1[4] = (unsigned)BM;                         // tile_dim1 = 128, tile_dim2 = 0
    g1[5] = (unsigned)K;                          // tensor_dim0_stride[31:0]
    g1[6] = 0u;                                   // stride[47:32]=0, dim1_stride lo = 0
    g1[7] = 0u;

    asm volatile("tensor_load_to_lds %0, %1" :: "s"(g0), "s"(g1) : "memory");
}

// ---------------------------------------------------------------------------
// Fused MoE GEMM: C(e) = epilogue( A(e)[MxK] * Bt(e)[NxK]^T + bias(e) )
// Both operands f16, K-major; staged by TDM into double-buffered padded LDS.
// ---------------------------------------------------------------------------
template<bool DOGELU>
__global__ __launch_bounds__(256)
void moe_gemm_tdm(const f16* __restrict__ A, const f16* __restrict__ Bt,
                  const float* __restrict__ biasp, void* __restrict__ Cp,
                  int K, int N)
{
    const int e    = blockIdx.z;
    const int n0   = blockIdx.x * BN;
    const int m0   = blockIdx.y * BM;
    const int tid  = threadIdx.x;
    const int lane = tid & 31;
    const int wave = tid >> 5;
    const int wm   = wave >> 1;   // 0..3 : 32-row strip
    const int wn   = wave & 1;    // 0..1 : 64-col strip

    // [buffer][A=0/B=1][row][padded K] ; single shared object -> LDS offset 0
    __shared__ __align__(16) f16 smem[2][2][BM][LDK];

    const f16* Abase = A  + (size_t)(e * TROW + m0) * K;
    const f16* Bbase = Bt + (size_t)e * N * K + (size_t)n0 * K;

    auto ldsoff = [](int buf, int op) -> unsigned {
        return (unsigned)(((buf * 2 + op) * BM * LDK) * (int)sizeof(f16));
    };

    v8f c[2][4];
#pragma unroll
    for (int mi = 0; mi < 2; ++mi)
#pragma unroll
        for (int ni = 0; ni < 4; ++ni)
            c[mi][ni] = {};

    const int KT = K / BK;

    if (wave == 0) {                               // TDM ignores EXEC; one wave issues
        tdm_tile_load(ldsoff(0, 0), Abase, K);
        tdm_tile_load(ldsoff(0, 1), Bbase, K);
    }

    for (int kt = 0; kt < KT; ++kt) {
        const int buf = kt & 1;
        if (wave == 0) {
            if (kt + 1 < KT) {                     // prefetch next stage into other buffer
                tdm_tile_load(ldsoff(buf ^ 1, 0), Abase + (size_t)(kt + 1) * BK, K);
                tdm_tile_load(ldsoff(buf ^ 1, 1), Bbase + (size_t)(kt + 1) * BK, K);
                __builtin_amdgcn_s_wait_tensorcnt(2);   // current pair complete (in-order)
            } else {
                __builtin_amdgcn_s_wait_tensorcnt(0);
            }
        }
        __syncthreads();                           // LDS tiles visible to all waves

#pragma unroll
        for (int kk = 0; kk < BK; kk += 32) {
            v16h a[2], b[4];
            // A fragment: lane L holds row M=L&15; K runs {ck..ck+7} and {ck+16..ck+23}
            const int arow = wm * 32 + (lane & 15);
            const int ck   = kk + (lane >> 4) * 8;
#pragma unroll
            for (int mi = 0; mi < 2; ++mi) {
                const f16* ap = &smem[buf][0][arow + mi * 16][0];
#pragma unroll
                for (int j = 0; j < 8; ++j) {
                    a[mi][j]     = ap[ck + j];
                    a[mi][8 + j] = ap[ck + 16 + j];
                }
            }
            // B fragment: lane L holds column N=L&15; K runs {kb..kb+7} and {kb+16..kb+23}
            const int brow = wn * 64 + (lane & 15);
            const int kb   = kk + (lane >> 4) * 8;
#pragma unroll
            for (int ni = 0; ni < 4; ++ni) {
                const f16* bp = &smem[buf][1][brow + ni * 16][0];
#pragma unroll
                for (int j = 0; j < 8; ++j) {
                    b[ni][j]     = bp[kb + j];
                    b[ni][8 + j] = bp[kb + 16 + j];
                }
            }
#pragma unroll
            for (int mi = 0; mi < 2; ++mi)
#pragma unroll
                for (int ni = 0; ni < 4; ++ni)
                    c[mi][ni] = __builtin_amdgcn_wmma_f32_16x16x32_f16(
                        false, a[mi], false, b[ni], (short)0, c[mi][ni], false, false);
        }
        __syncthreads();                           // done reading buf before overwrite
    }

    // Epilogue: bias (+ exact-erf GELU). C layout: VGPR r -> M = r + 8*(lane>=16)
    const float* bias = biasp + (size_t)e * N;
#pragma unroll
    for (int mi = 0; mi < 2; ++mi) {
#pragma unroll
        for (int ni = 0; ni < 4; ++ni) {
            const int ncol = n0 + wn * 64 + ni * 16 + (lane & 15);
            const float bv = bias[ncol];
#pragma unroll
            for (int r = 0; r < 8; ++r) {
                const int mrow = m0 + wm * 32 + mi * 16 + (lane >> 4) * 8 + r;
                float v = c[mi][ni][r] + bv;
                const size_t idx = (size_t)(e * TROW + mrow) * N + ncol;
                if constexpr (DOGELU) {
                    v = 0.5f * v * (1.0f + erff(v * 0.70710678118654752f));
                    ((f16*)Cp)[idx] = (f16)v;
                } else {
                    ((float*)Cp)[idx] = v;
                }
            }
        }
    }
}

extern "C" void kernel_launch(void* const* d_in, const int* in_sizes, int n_in,
                              void* d_out, int out_size, void* d_ws, size_t ws_size,
                              hipStream_t stream) {
    (void)in_sizes; (void)n_in; (void)out_size; (void)ws_size;
    const float* x  = (const float*)d_in[0];   // (E*T, D) fp32 (flat order)
    const float* w1 = (const float*)d_in[1];   // (E, D, H)
    const float* b1 = (const float*)d_in[2];   // (E, H)
    const float* w2 = (const float*)d_in[3];   // (E, H, D)
    const float* b2 = (const float*)d_in[4];   // (E, D)
    float* out = (float*)d_out;                // (E*T, D) fp32

    // Workspace layout (f16), requires >= 256 MiB:
    //   x16  : E*T*D   = 33.5M halves (64 MB)
    //   w1t  : E*H*D   = 16.8M halves (32 MB)  (per-expert H x D, K-major)
    //   w2t  : E*D*H   = 16.8M halves (32 MB)  (per-expert D x H, K-major)
    //   hact : E*T*H   = 67.1M halves (128 MB)
    const long long XN  = (long long)NEXP * TROW * DDIM;
    const long long W1N = (long long)NEXP * DDIM * HDIM;
    const long long W2N = (long long)NEXP * HDIM * DDIM;
    f16* x16 = (f16*)d_ws;
    f16* w1t = x16 + XN;
    f16* w2t = w1t + W1N;
    f16* hws = w2t + W2N;

    dim3 blk(256);

    // Prep: convert + transpose (one-time, bandwidth-trivial vs compute)
    cvt_f32_to_f16<<<dim3((unsigned)(XN / 2048)), blk, 0, stream>>>(x, x16, XN);
    transpose_cvt_f16<<<dim3(HDIM / 32, DDIM / 32, NEXP), blk, 0, stream>>>(w1, w1t, DDIM, HDIM);
    transpose_cvt_f16<<<dim3(DDIM / 32, HDIM / 32, NEXP), blk, 0, stream>>>(w2, w2t, HDIM, DDIM);

    // GEMM1: h = gelu(x @ w1 + b1)   (A: E*T x D, B^T: H x D)
    moe_gemm_tdm<true ><<<dim3(HDIM / BN, TROW / BM, NEXP), blk, 0, stream>>>(x16, w1t, b1, hws, DDIM, HDIM);
    // GEMM2: out = h @ w2 + b2       (A: E*T x H, B^T: D x H)
    moe_gemm_tdm<false><<<dim3(DDIM / BN, TROW / BM, NEXP), blk, 0, stream>>>(hws, w2t, b2, out, HDIM, DDIM);
}
